// PureInterferenceAttention_16827681866026
// MI455X (gfx1250) — matrix operand order repaired and checked by reading, not compiled, not run on hardware
//
#include <hip/hip_runtime.h>

typedef __attribute__((ext_vector_type(16))) _Float16 v16h;
typedef __attribute__((ext_vector_type(8)))  _Float16 v8h;
typedef __attribute__((ext_vector_type(8)))  float    v8f;

#define D_MODEL   1024
#define N_HEADS   16
#define N_PHASE   128
#define HEAD_DIM  64
#define SEQ_B     2
#define SEQ_L     2048
#define N_TOK     (SEQ_B * SEQ_L)          // 4096
#define SCALE_ATTN 0.35355339059327373f    // 1/sqrt(8)
#define LN_EPS    1e-5f
#define LOG1E4_64 0.14391156831209823f     // ln(10000)/64

// ---------------------------------------------------------------------------
// WMMA fragment loaders (gfx1250 16x16x32 f16 layouts, wave32)
// A (16xK=32): lane m=lane&15, h=lane>>4; VGPR0-3 hold K=8h..8h+7,
//              VGPR4-7 hold K=16+8h..16+8h+7  -> two 16B loads.
// B (K=32x16): lane n=lane&15, h=lane>>4; same K order as A -> two 16B loads.
// C/D f32:     m = r + 8*(lane>>4), n = lane&15 for register r=0..7.
// ---------------------------------------------------------------------------
__device__ static inline v16h load_frag_a(const _Float16* __restrict__ base,
                                          int row0, int ld, int k0) {
  int lane = threadIdx.x & 31;
  int m = lane & 15, hs = lane >> 4;
  const _Float16* p = base + (size_t)(row0 + m) * ld + k0;
  v8h lo = *(const v8h*)(p + 8 * hs);
  v8h hi = *(const v8h*)(p + 16 + 8 * hs);
  v16h r;
#pragma unroll
  for (int i = 0; i < 8; ++i) { r[i] = lo[i]; r[8 + i] = hi[i]; }
  return r;
}

// Loads B[k][n] = src[n0+n][k0+k] (rows of src are the N dimension).
__device__ static inline v16h load_frag_b(const _Float16* __restrict__ base,
                                          int n0, int ld, int k0) {
  int lane = threadIdx.x & 31;
  int n = lane & 15, hs = lane >> 4;
  const _Float16* p = base + (size_t)(n0 + n) * ld + k0;
  v8h lo = *(const v8h*)(p + 8 * hs);
  v8h hi = *(const v8h*)(p + 16 + 8 * hs);
  v16h r;
#pragma unroll
  for (int i = 0; i < 8; ++i) { r[i] = lo[i]; r[8 + i] = hi[i]; }
  return r;
}

// Per-lane prefetch of the A tile (one address per lane covers the 16 rows).
__device__ static inline void prefetch_a(const _Float16* __restrict__ base,
                                         int row0, int ld, int k0) {
  int lane = threadIdx.x & 31;
  const _Float16* p = base + (size_t)(row0 + (lane & 15)) * ld + k0 +
                      32 * (lane >> 4);
  __builtin_prefetch(p, 0, 3);   // -> global_prefetch_b8, near scope
}

__device__ static inline v8f wmma_f16(v16h a, v16h b, v8f c) {
  return __builtin_amdgcn_wmma_f32_16x16x32_f16(false, a, false, b,
                                                (short)0, c, false, false);
}

// ---------------------------------------------------------------------------
// Elementwise helpers
// ---------------------------------------------------------------------------
__global__ void cvt_f32_f16(const float* __restrict__ in,
                            _Float16* __restrict__ out, int n) {
  int i = blockIdx.x * blockDim.x + threadIdx.x;
  int stride = gridDim.x * blockDim.x;
  for (; i < n; i += stride) out[i] = (_Float16)in[i];
}

__global__ void zero_f16(_Float16* __restrict__ out, int n) {
  int i = blockIdx.x * blockDim.x + threadIdx.x;
  int stride = gridDim.x * blockDim.x;
  for (; i < n; i += stride) out[i] = (_Float16)0.0f;
}

__global__ void copy_f32(const float* __restrict__ in,
                         float* __restrict__ out, int n) {
  int i = blockIdx.x * blockDim.x + threadIdx.x;
  int stride = gridDim.x * blockDim.x;
  for (; i < n; i += stride) out[i] = in[i];
}

// ---------------------------------------------------------------------------
// GEMM inner step: all fragment loads issued before the WMMA chain so the
// backend can use partial s_wait_loadcnt values instead of draining to 0.
// ---------------------------------------------------------------------------
__device__ static inline void gemm_step_16x64(const _Float16* __restrict__ A,
                                              const _Float16* __restrict__ B,
                                              int m0, int n0, int k0,
                                              v8f acc[4]) {
  // prefetch 2 k-steps ahead (branchless clamp keeps it in-bounds)
  int kp = k0 + 64;
  kp = (kp < D_MODEL) ? kp : (D_MODEL - 32);
  prefetch_a(A, m0, D_MODEL, kp);
  v16h a = load_frag_a(A, m0, D_MODEL, k0);
  v16h bf[4];
#pragma unroll
  for (int t = 0; t < 4; ++t) bf[t] = load_frag_b(B, n0 + 16 * t, D_MODEL, k0);
#pragma unroll
  for (int t = 0; t < 4; ++t) acc[t] = wmma_f16(a, bf[t], acc[t]);
}

// ---------------------------------------------------------------------------
// Q/K phase projection: Phase = x_imag @ W^T + bias + pos; emit cos/sin
// features into feat[b][h][l][32] (f16, cols 0..7 cos, 8..15 sin, 16..31 = 0)
// Each wave computes a 16x64 tile. grid (N_TOK/16, N_PHASE/64), block 32
// ---------------------------------------------------------------------------
__global__ void qk_phase(const _Float16* __restrict__ X,
                         const _Float16* __restrict__ W,
                         const float* __restrict__ bias,
                         _Float16* __restrict__ feat) {
  int m0 = blockIdx.x * 16;
  int n0 = blockIdx.y * 64;
  v8f zero = {0.f, 0.f, 0.f, 0.f, 0.f, 0.f, 0.f, 0.f};
  v8f acc[4];
#pragma unroll
  for (int t = 0; t < 4; ++t) acc[t] = zero;
#pragma unroll 2
  for (int k0 = 0; k0 < D_MODEL; k0 += 32)
    gemm_step_16x64(X, W, m0, n0, k0, acc);

  int lane = threadIdx.x & 31;
  int hs = lane >> 4;
#pragma unroll
  for (int t = 0; t < 4; ++t) {
    int n = n0 + 16 * t + (lane & 15);
    int head = n >> 3, f = n & 7;
    float invf = __expf(-LOG1E4_64 * (float)(n >> 1));
    float bn = bias[n];
#pragma unroll
    for (int r = 0; r < 8; ++r) {
      int m = m0 + r + 8 * hs;
      int b_ = m >> 11, l = m & (SEQ_L - 1);
      float phase = acc[t][r] + bn + (float)l * invf;
      size_t o = ((size_t)((b_ * N_HEADS + head) * SEQ_L + l)) * 32 + f;
      feat[o]     = (_Float16)__cosf(phase);
      feat[o + 8] = (_Float16)__sinf(phase);
    }
  }
}

// ---------------------------------------------------------------------------
// V = x_real @ Wv^T, stored transposed: Vt[b][h][d][l]  (f16)
// 16x64 tile per wave. grid (N_TOK/16, D_MODEL/64), block 32
// ---------------------------------------------------------------------------
__global__ void v_proj(const _Float16* __restrict__ X,
                       const _Float16* __restrict__ Wv,
                       _Float16* __restrict__ Vt) {
  int m0 = blockIdx.x * 16;
  int n0 = blockIdx.y * 64;
  v8f zero = {0.f, 0.f, 0.f, 0.f, 0.f, 0.f, 0.f, 0.f};
  v8f acc[4];
#pragma unroll
  for (int t = 0; t < 4; ++t) acc[t] = zero;
#pragma unroll 2
  for (int k0 = 0; k0 < D_MODEL; k0 += 32)
    gemm_step_16x64(X, Wv, m0, n0, k0, acc);

  int lane = threadIdx.x & 31;
  int hs = lane >> 4;
#pragma unroll
  for (int t = 0; t < 4; ++t) {
    int n = n0 + 16 * t + (lane & 15);
    int head = n >> 6, d = n & 63;
#pragma unroll
    for (int r = 0; r < 8; ++r) {
      int m = m0 + r + 8 * hs;
      int b_ = m >> 11, l = m & (SEQ_L - 1);
      Vt[((size_t)(b_ * N_HEADS + head) * HEAD_DIM + d) * SEQ_L + l] =
          (_Float16)acc[t][r];
    }
  }
}

// ---------------------------------------------------------------------------
// Fused flash attention (causal). One wave per (b, h, 16-row tile).
// 64 keys per iteration: 4 score WMMAs (shared Q fragment, K=32 padded
// feature dim), online softmax with intra-half shuffle reductions, P tile
// (16x64) transposed through LDS, 8 PV WMMAs.
// grid (SEQ_L/16, N_HEADS, SEQ_B), block 32
// ---------------------------------------------------------------------------
__global__ void attn_fused(const _Float16* __restrict__ qf,
                           const _Float16* __restrict__ kf,
                           const _Float16* __restrict__ Vt,
                           _Float16* __restrict__ AO) {
  __shared__ alignas(16) _Float16 lp[16 * 64];
  int m0 = blockIdx.x * 16;
  int h  = blockIdx.y;
  int b  = blockIdx.z;
  const _Float16* qb = qf + (size_t)(b * N_HEADS + h) * SEQ_L * 32;
  const _Float16* kb = kf + (size_t)(b * N_HEADS + h) * SEQ_L * 32;
  const _Float16* vb = Vt + (size_t)(b * N_HEADS + h) * HEAD_DIM * SEQ_L;

  int lane = threadIdx.x & 31;
  int n = lane & 15, hs = lane >> 4;

  v16h aq = load_frag_a(qb, m0, 32, 0);
  v8f zero = {0.f, 0.f, 0.f, 0.f, 0.f, 0.f, 0.f, 0.f};
  float Mx[8], Ls[8];
  v8f acc[4];
#pragma unroll
  for (int r = 0; r < 8; ++r) { Mx[r] = -3.0e38f; Ls[r] = 0.f; }
#pragma unroll
  for (int t = 0; t < 4; ++t) acc[t] = zero;

  int jend = m0 + 16;  // causal: max valid column index is m0+15
  for (int jc = 0; jc < jend; jc += 64) {
    v16h kfr[4];
#pragma unroll
    for (int t = 0; t < 4; ++t)
      kfr[t] = load_frag_b(kb, jc + 16 * t, 32, 0);   // loads first
    v8f s[4];
#pragma unroll
    for (int t = 0; t < 4; ++t) s[t] = wmma_f16(aq, kfr[t], zero);
#pragma unroll
    for (int r = 0; r < 8; ++r) {
      int row = m0 + r + 8 * hs;
      float pv[4];
      float rm = -3.0e38f;
#pragma unroll
      for (int t = 0; t < 4; ++t) {
        pv[t] = (jc + 16 * t + n <= row) ? s[t][r] * SCALE_ATTN : -1.0e30f;
        rm = fmaxf(rm, pv[t]);
      }
      rm = fmaxf(rm, __shfl_xor(rm, 1, 32));   // row max (16 lanes per half)
      rm = fmaxf(rm, __shfl_xor(rm, 2, 32));
      rm = fmaxf(rm, __shfl_xor(rm, 4, 32));
      rm = fmaxf(rm, __shfl_xor(rm, 8, 32));
      float newm = fmaxf(Mx[r], rm);
      float sc = __expf(Mx[r] - newm);
      Mx[r] = newm;
      float rs = 0.f;
      int ml = r + 8 * hs;                     // C-layout row -> LDS rowmajor
#pragma unroll
      for (int t = 0; t < 4; ++t) {
        float p = __expf(pv[t] - newm);
        rs += p;
        lp[ml * 64 + 16 * t + n] = (_Float16)p;
      }
      rs += __shfl_xor(rs, 1, 32);             // row sum
      rs += __shfl_xor(rs, 2, 32);
      rs += __shfl_xor(rs, 4, 32);
      rs += __shfl_xor(rs, 8, 32);
      Ls[r] = Ls[r] * sc + rs;
#pragma unroll
      for (int t = 0; t < 4; ++t) acc[t][r] *= sc;
    }
    __syncthreads();
    v16h pf[2];  // reload P (16x64) as two K=32 A-fragments
#pragma unroll
    for (int kk = 0; kk < 2; ++kk) {
      const _Float16* pr = &lp[(lane & 15) * 64 + 32 * kk];
      v8h lo = *(const v8h*)(pr + 8 * hs);
      v8h hi = *(const v8h*)(pr + 16 + 8 * hs);
#pragma unroll
      for (int i = 0; i < 8; ++i) { pf[kk][i] = lo[i]; pf[kk][8 + i] = hi[i]; }
    }
#pragma unroll
    for (int kk = 0; kk < 2; ++kk) {
      v16h vfr[4];
#pragma unroll
      for (int t = 0; t < 4; ++t)
        vfr[t] = load_frag_b(vb, t * 16, SEQ_L, jc + 32 * kk);  // loads first
#pragma unroll
      for (int t = 0; t < 4; ++t)
        acc[t] = wmma_f16(pf[kk], vfr[t], acc[t]);
    }
    __syncthreads();
  }

#pragma unroll
  for (int r = 0; r < 8; ++r) {
    int l = m0 + r + 8 * hs;
    float inv = 1.0f / Ls[r];
    size_t ro = ((size_t)(b * SEQ_L + l)) * D_MODEL + h * HEAD_DIM;
#pragma unroll
    for (int t = 0; t < 4; ++t)
      AO[ro + t * 16 + n] = (_Float16)(acc[t][r] * inv);
  }
}

// ---------------------------------------------------------------------------
// Output projection + residual: H = AO @ Wo^T + x_real  (f32 into d_out)
// 16x64 tile per wave. grid (N_TOK/16, D_MODEL/64), block 32
// ---------------------------------------------------------------------------
__global__ void out_proj(const _Float16* __restrict__ AO,
                         const _Float16* __restrict__ Wo,
                         const float* __restrict__ xr,
                         float* __restrict__ H) {
  int m0 = blockIdx.x * 16;
  int n0 = blockIdx.y * 64;
  v8f zero = {0.f, 0.f, 0.f, 0.f, 0.f, 0.f, 0.f, 0.f};
  v8f acc[4];
#pragma unroll
  for (int t = 0; t < 4; ++t) acc[t] = zero;
#pragma unroll 2
  for (int k0 = 0; k0 < D_MODEL; k0 += 32)
    gemm_step_16x64(AO, Wo, m0, n0, k0, acc);

  int lane = threadIdx.x & 31;
  int hs = lane >> 4;
#pragma unroll
  for (int t = 0; t < 4; ++t) {
    int n = n0 + 16 * t + (lane & 15);
#pragma unroll
    for (int r = 0; r < 8; ++r) {
      int m = m0 + r + 8 * hs;
      size_t o = (size_t)m * D_MODEL + n;
      H[o] = acc[t][r] + xr[o];
    }
  }
}

// ---------------------------------------------------------------------------
// In-place LayerNorm over rows of 1024. grid N_TOK, block 256.
// ---------------------------------------------------------------------------
__global__ void layernorm_inplace(float* __restrict__ H,
                                  const float* __restrict__ gamma,
                                  const float* __restrict__ beta) {
  __shared__ float s1[256];
  __shared__ float s2[256];
  int row = blockIdx.x;
  float* p = H + (size_t)row * D_MODEL;
  int tid = threadIdx.x;
  float v[4];
  float a = 0.f, bsum = 0.f;
#pragma unroll
  for (int i = 0; i < 4; ++i) {
    v[i] = p[tid + i * 256];
    a += v[i];
    bsum += v[i] * v[i];
  }
  s1[tid] = a; s2[tid] = bsum;
  __syncthreads();
  for (int off = 128; off > 0; off >>= 1) {
    if (tid < off) { s1[tid] += s1[tid + off]; s2[tid] += s2[tid + off]; }
    __syncthreads();
  }
  float mean = s1[0] * (1.0f / D_MODEL);
  float var  = s2[0] * (1.0f / D_MODEL) - mean * mean;
  float inv  = rsqrtf(var + LN_EPS);
#pragma unroll
  for (int i = 0; i < 4; ++i) {
    int c = tid + i * 256;
    p[c] = (v[i] - mean) * inv * gamma[c] + beta[c];
  }
}

// ---------------------------------------------------------------------------
extern "C" void kernel_launch(void* const* d_in, const int* in_sizes, int n_in,
                              void* d_out, int out_size, void* d_ws,
                              size_t ws_size, hipStream_t stream) {
  (void)in_sizes; (void)n_in; (void)out_size; (void)ws_size;
  const float* x_real = (const float*)d_in[0];
  const float* x_imag = (const float*)d_in[1];
  const float* Wq     = (const float*)d_in[2];
  const float* bq     = (const float*)d_in[3];
  const float* Wk     = (const float*)d_in[4];
  const float* bk     = (const float*)d_in[5];
  const float* Wv     = (const float*)d_in[6];
  const float* Wo     = (const float*)d_in[7];
  const float* gamma  = (const float*)d_in[8];
  const float* beta   = (const float*)d_in[9];
  float* out = (float*)d_out;

  char* ws = (char*)d_ws;                     // ~45 MB used
  _Float16* xi_h = (_Float16*)(ws + ((size_t)0  << 20));
  _Float16* xr_h = (_Float16*)(ws + ((size_t)8  << 20));
  _Float16* Wq_h = (_Float16*)(ws + ((size_t)16 << 20));
  _Float16* Wk_h = (_Float16*)(ws + ((size_t)16 << 20) + (256u << 10));
  _Float16* Wv_h = (_Float16*)(ws + ((size_t)17 << 20));
  _Float16* Wo_h = (_Float16*)(ws + ((size_t)19 << 20));
  _Float16* qf   = (_Float16*)(ws + ((size_t)21 << 20));
  _Float16* kf   = (_Float16*)(ws + ((size_t)25 << 20));
  _Float16* Vt   = (_Float16*)(ws + ((size_t)29 << 20));
  _Float16* AO   = (_Float16*)(ws + ((size_t)37 << 20));

  const int NT = N_TOK * D_MODEL;             // 4194304
  cvt_f32_f16<<<1024, 256, 0, stream>>>(x_imag, xi_h, NT);
  cvt_f32_f16<<<1024, 256, 0, stream>>>(x_real, xr_h, NT);
  cvt_f32_f16<<<128,  256, 0, stream>>>(Wq, Wq_h, N_PHASE * D_MODEL);
  cvt_f32_f16<<<128,  256, 0, stream>>>(Wk, Wk_h, N_PHASE * D_MODEL);
  cvt_f32_f16<<<1024, 256, 0, stream>>>(Wv, Wv_h, D_MODEL * D_MODEL);
  cvt_f32_f16<<<1024, 256, 0, stream>>>(Wo, Wo_h, D_MODEL * D_MODEL);

  const int NF = SEQ_B * N_HEADS * SEQ_L * 32;  // feature buffers (zero pad)
  zero_f16<<<512, 256, 0, stream>>>(qf, NF);
  zero_f16<<<512, 256, 0, stream>>>(kf, NF);

  dim3 gq(N_TOK / 16, N_PHASE / 64, 1);
  qk_phase<<<gq, 32, 0, stream>>>(xi_h, Wq_h, bq, qf);
  qk_phase<<<gq, 32, 0, stream>>>(xi_h, Wk_h, bk, kf);

  dim3 gv(N_TOK / 16, D_MODEL / 64, 1);
  v_proj<<<gv, 32, 0, stream>>>(xr_h, Wv_h, Vt);

  dim3 ga(SEQ_L / 16, N_HEADS, SEQ_B);
  attn_fused<<<ga, 32, 0, stream>>>(qf, kf, Vt, AO);

  dim3 go(N_TOK / 16, D_MODEL / 64, 1);
  out_proj<<<go, 32, 0, stream>>>(AO, Wo_h, x_real, out);

  layernorm_inplace<<<N_TOK, 256, 0, stream>>>(out, gamma, beta);

  copy_f32<<<1024, 256, 0, stream>>>(x_imag, out + (size_t)N_TOK * D_MODEL, NT);
}
